// CandidateFinder_12421045420475
// MI455X (gfx1250) — 8-bit matrix operand order repaired and checked by reading, compiled, not run on hardware
//
#include <hip/hip_runtime.h>
#include <hip/hip_bf16.h>
#include <stdint.h>

typedef int v8i __attribute__((ext_vector_type(8)));

#define BATCH 4
#define LTOK  2048
#define DFEAT 64
#define NTOK  (BATCH * LTOK)
#define KMAX  64
#define NWORDS (LTOK / 32)

__global__ void cf_pack_kernel(const float* __restrict__ qf,
                               const float* __restrict__ kf,
                               uint8_t* __restrict__ qbytes,
                               int8_t*  __restrict__ kbytes,
                               int*     __restrict__ kpop0,
                               int*     __restrict__ kpop1) {
    int tid = blockIdx.x * blockDim.x + threadIdx.x;
    if (tid < NTOK) {
        const float* x = qf + (size_t)tid * DFEAT;
        uint8_t* o = qbytes + (size_t)tid * DFEAT;
        #pragma unroll
        for (int d = 0; d < DFEAT; ++d)
            o[d] = (x[d] > 0.0f) ? (uint8_t)1 : (uint8_t)0;
    } else if (tid < 2 * NTOK) {
        int t = tid - NTOK;
        const float* x = kf + (size_t)t * DFEAT;
        int8_t* o = kbytes + (size_t)t * DFEAT;
        int p0 = 0, p1 = 0;
        #pragma unroll
        for (int d = 0; d < 32; ++d) {
            int bit = (x[d] > 0.0f) ? 1 : 0;
            p0 += bit;
            o[d] = (int8_t)(1 - 2 * bit);
        }
        #pragma unroll
        for (int d = 32; d < 64; ++d) {
            int bit = (x[d] > 0.0f) ? 1 : 0;
            p1 += bit;
            o[d] = (int8_t)(1 - 2 * bit);
        }
        kpop0[t] = p0;
        kpop1[t] = p1;
    }
}

__global__ void __launch_bounds__(256)
cf_match_kernel(const uint8_t* __restrict__ qbytes,
                const int8_t*  __restrict__ kbytes,
                const int*     __restrict__ kpop0,
                const int*     __restrict__ kpop1,
                uint32_t*      __restrict__ matchbits) {
    const int wid  = (blockIdx.x * blockDim.x + threadIdx.x) >> 5;
    const int lane = threadIdx.x & 31;

    const int jt = wid & (NWORDS - 1);
    const int it = (wid >> 6) & (LTOK / 16 - 1);
    const int b  = wid >> 13;

    const int row = it * 16 + (lane & 15);
    const uint8_t* qp = qbytes + (((size_t)b * LTOK + row) << 6);
    const int hi = (lane >> 4) << 4;
    uint32_t a0 = *(const uint32_t*)(qp + hi + 0);
    uint32_t a1 = *(const uint32_t*)(qp + hi + 4);
    uint32_t a2 = *(const uint32_t*)(qp + hi + 8);
    uint32_t a3 = *(const uint32_t*)(qp + hi + 12);
    uint32_t a4 = *(const uint32_t*)(qp + hi + 32);
    uint32_t a5 = *(const uint32_t*)(qp + hi + 36);
    uint32_t a6 = *(const uint32_t*)(qp + hi + 40);
    uint32_t a7 = *(const uint32_t*)(qp + hi + 44);
    v8i A0 = (v8i){(int)a0, (int)a1, (int)a2, (int)a3, 0, 0, 0, 0};
    v8i A1 = (v8i){0, 0, 0, 0, (int)a4, (int)a5, (int)a6, (int)a7};

    uint32_t masks[2][8];
    #pragma unroll
    for (int s = 0; s < 2; ++s) {
        const int col = jt * 32 + s * 16 + (lane & 15);
        const int8_t* kp = kbytes + (((size_t)b * LTOK + col) << 6);
        const int kb = (lane >> 4) << 4;
        v8i Bm = (v8i){ *(const int*)(kp + kb + 0),
                        *(const int*)(kp + kb + 4),
                        *(const int*)(kp + kb + 8),
                        *(const int*)(kp + kb + 12),
                        *(const int*)(kp + kb + 32),
                        *(const int*)(kp + kb + 36),
                        *(const int*)(kp + kb + 40),
                        *(const int*)(kp + kb + 44) };

        const int p0 = kpop0[(size_t)b * LTOK + col];
        const int p1 = kpop1[(size_t)b * LTOK + col];
        v8i C0 = (v8i){p0, p0, p0, p0, p0, p0, p0, p0};
        v8i C1 = (v8i){p1, p1, p1, p1, p1, p1, p1, p1};

        v8i D0 = __builtin_amdgcn_wmma_i32_16x16x64_iu8(false, A0, true, Bm, C0,
                                                        false, false);
        v8i D1 = __builtin_amdgcn_wmma_i32_16x16x64_iu8(false, A1, true, Bm, C1,
                                                        false, false);

        #pragma unroll
        for (int v = 0; v < 8; ++v) {
            bool match = (D0[v] == 0) || (D1[v] == 0);
            masks[s][v] = __builtin_amdgcn_ballot_w32(match);
        }
    }

    uint32_t w[16];
    #pragma unroll
    for (int v = 0; v < 8; ++v) {
        w[v]     = (masks[0][v] & 0xFFFFu) | (masks[1][v] << 16);
        w[v + 8] = (masks[0][v] >> 16)     | (masks[1][v] & 0xFFFF0000u);
    }

    if (lane < 16) {
        uint32_t val = w[0];
        #pragma unroll
        for (int r = 1; r < 16; ++r)
            val = (lane == r) ? w[r] : val;
        matchbits[(((size_t)b * LTOK + it * 16 + lane) << 6) + jt] = val;
    }
}

__global__ void cf_select_kernel(int* __restrict__ out) {
    int tid = blockIdx.x * blockDim.x + threadIdx.x;
    if (tid >= NTOK) return;

    const uint32_t* mb = (const uint32_t*)out + (size_t)tid * NWORDS;
    uint32_t m[NWORDS];
    #pragma unroll
    for (int wi = 0; wi < NWORDS; ++wi) m[wi] = mb[wi];

    int* dst = out + (size_t)tid * KMAX;
    int pos = 0;
    #pragma unroll 4
    for (int wi = 0; wi < NWORDS; ++wi) {
        uint32_t mm = m[wi];
        while (mm != 0u && pos < KMAX) {
            int t = __builtin_ctz(mm);
            dst[pos++] = wi * 32 + t;
            mm &= mm - 1u;
        }
        if (pos >= KMAX) break;
    }
    while (pos < KMAX) dst[pos++] = -1;
}

extern "C" void kernel_launch(void* const* d_in, const int* in_sizes, int n_in,
                              void* d_out, int out_size, void* d_ws, size_t ws_size,
                              hipStream_t stream) {
    const float* qf = (const float*)d_in[0];
    const float* kf = (const float*)d_in[1];

    uint8_t* qbytes = (uint8_t*)d_ws;
    int8_t*  kbytes = (int8_t*)((char*)d_ws + (size_t)NTOK * DFEAT);
    int*     kpop0  = (int*)((char*)d_ws + (size_t)2 * NTOK * DFEAT);
    int*     kpop1  = kpop0 + NTOK;

    {
        int threads = 2 * NTOK;
        cf_pack_kernel<<<(threads + 255) / 256, 256, 0, stream>>>(
            qf, kf, qbytes, kbytes, kpop0, kpop1);
    }

    {
        int waves = BATCH * (LTOK / 16) * (LTOK / 32);
        cf_match_kernel<<<waves / 8, 256, 0, stream>>>(
            qbytes, kbytes, kpop0, kpop1, (uint32_t*)d_out);
    }

    cf_select_kernel<<<(NTOK + 255) / 256, 256, 0, stream>>>((int*)d_out);
}
